// MixingAttention_85152021610850
// MI455X (gfx1250) — compile-verified, matrix operand order checked by reading, not run on hardware
//
#include <hip/hip_runtime.h>
#include <hip/hip_bf16.h>
#include <math.h>

typedef __attribute__((ext_vector_type(16))) _Float16 v16h;
typedef __attribute__((ext_vector_type(8)))  _Float16 v8h;
typedef __attribute__((ext_vector_type(8)))  float    v8f;

#define WSZ     8
#define NHEADS  8
#define CDIM    512
#define C2DIM   256
#define C8DIM   64
#define HDIM    32
#define NTOK    64          // WSZ*WSZ
#define BATCH   2
#define HIMG    224
#define WIMG    224
#define NWIN_X  28
#define NWIN    784         // windows per image
#define BWIN    1568        // total windows (B_)
#define MROWS   (BWIN*NTOK) // 100352
#define HWPIX   (HIMG*WIMG) // 50176
#define EPSLN   1e-5f

#define BM 128
#define BK 32
#define ASTR 40   // LDS row stride (halfs) -> conflict-free ds_load_b128
#define BSTR 40

// ---- optional CDNA5 async global->LDS path (probe-guarded) ----
#if __has_builtin(__builtin_amdgcn_global_load_async_to_lds_b128)
#define HAVE_ASYNC_LDS 1
#else
#define HAVE_ASYNC_LDS 0
#endif

#if HAVE_ASYNC_LDS
typedef int v4i_async __attribute__((vector_size(4 * sizeof(int))));
typedef v4i_async __attribute__((address_space(1)))* async_gptr;
typedef v4i_async __attribute__((address_space(3)))* async_lptr;

__device__ __forceinline__ void async_cp16(const _Float16* g, _Float16* l) {
    __builtin_amdgcn_global_load_async_to_lds_b128(
        (async_gptr)(g), (async_lptr)(l), 0, 0);
}
__device__ __forceinline__ void wait_async() {
#if __has_builtin(__builtin_amdgcn_s_wait_asynccnt)
    __builtin_amdgcn_s_wait_asynccnt(0);
#else
    asm volatile("s_wait_asynccnt 0x0" ::: "memory");
#endif
}
#endif

__device__ __forceinline__ float gelu_f(float x) {
    return 0.5f * x * (1.0f + erff(x * 0.70710678118654752f));
}
__device__ __forceinline__ float sigmoid_f(float x) {
    return 1.0f / (1.0f + __expf(-x));
}
// 16x32 f16 WMMA fragment from LDS: per-lane row = base + lane%16,
// K pattern kb = (lane>>4)*8 covering {kb..kb+7, kb+16..kb+23}.
__device__ __forceinline__ v16h frag_ld(const _Float16* p) {
    v8h lo = *(const v8h*)p;
    v8h hi = *(const v8h*)(p + 16);
    v16h r;
#pragma unroll
    for (int e = 0; e < 8; ++e) { r[e] = lo[e]; r[e + 8] = hi[e]; }
    return r;
}

// ---------------- Generic GEMM: C[MxN](f32) = A[MxK](f16) * B[KxN](f32) + bias ----------
// A f16 in HBM (copied raw to LDS, async path if available); B converted f32->f16 while
// staging transposed+k-pair-packed; f32 accumulate via v_wmma_f32_16x16x32_f16.
// Grid: x = N/BNT (few), y = M/BM (784) so A-sharing blocks are launch-adjacent (L2 reuse).
template <int BNT>
__global__ __launch_bounds__(256) void gemm_f16wmma(
    const _Float16* __restrict__ A, const float* __restrict__ B,
    const float* __restrict__ bias, float* __restrict__ C,
    int M, int N, int K)
{
    __shared__ __attribute__((aligned(16))) _Float16 Al[BM * ASTR];
    __shared__ __attribute__((aligned(16))) _Float16 Bl[BNT * BSTR];

    const int t = threadIdx.x;
    const int lane = t & 31, wave = t >> 5;
    const int wm = (wave & 3) * 32;            // 4 waves along M (2 x 16-row tiles each)
    const int wn = (wave >> 2) * (BNT / 2);    // 2 waves along N
    const int lm = lane & 15;
    const int kb = (lane >> 4) * 8;
    const int col0 = blockIdx.x * BNT, row0 = blockIdx.y * BM;
    constexpr int NT = BNT / 32;               // 16-col tiles per wave

    v8f acc[2][NT] = {};

    // staging coordinates
    const int ar = t >> 1;            // 0..127
    const int ac = (t & 1) * 16;      // 0 / 16
    const int tk = (t & 15) * 2;      // k pair base 0..30
    const int tn0 = (t >> 4) * (BNT / 16);

    for (int k0 = 0; k0 < K; k0 += BK) {
        // ---- stage A tile 128x32 halfs (raw copy, 32B per thread) ----
        {
            const _Float16* gp = A + (size_t)(row0 + ar) * K + k0 + ac;
            _Float16* lp = &Al[ar * ASTR + ac];
#if HAVE_ASYNC_LDS
            async_cp16(gp, lp);
            async_cp16(gp + 8, lp + 8);
#else
            *(v8h*)lp = *(const v8h*)gp;
            *(v8h*)(lp + 8) = *(const v8h*)(gp + 8);
#endif
            if (k0 + BK < K)  // global_prefetch_b8 for next K tile
                __builtin_prefetch(A + (size_t)(row0 + ar) * K + k0 + BK + ac, 0, 1);
        }
        // ---- stage B tile 32xBNT f32 -> f16, transposed Bl[n][k], k-pairs packed b32 ----
        {
#pragma unroll
            for (int nn = 0; nn < BNT / 16; nn += 4) {
                const float4 f0 = *(const float4*)(B + (size_t)(k0 + tk) * N + col0 + tn0 + nn);
                const float4 f1 = *(const float4*)(B + (size_t)(k0 + tk + 1) * N + col0 + tn0 + nn);
                const float a0[4] = { f0.x, f0.y, f0.z, f0.w };
                const float a1[4] = { f1.x, f1.y, f1.z, f1.w };
#pragma unroll
                for (int i = 0; i < 4; ++i) {
                    union { _Float16 h[2]; unsigned u; } pk;
                    pk.h[0] = (_Float16)a0[i];
                    pk.h[1] = (_Float16)a1[i];
                    *(unsigned*)&Bl[(tn0 + nn + i) * BSTR + tk] = pk.u;
                }
            }
        }
#if HAVE_ASYNC_LDS
        wait_async();
#endif
        __syncthreads();

        v16h af[2], bf[NT];
#pragma unroll
        for (int i = 0; i < 2; ++i) af[i] = frag_ld(&Al[(wm + i * 16 + lm) * ASTR + kb]);
#pragma unroll
        for (int j = 0; j < NT; ++j) bf[j] = frag_ld(&Bl[(wn + j * 16 + lm) * BSTR + kb]);
#pragma unroll
        for (int i = 0; i < 2; ++i)
#pragma unroll
            for (int j = 0; j < NT; ++j)
                acc[i][j] = __builtin_amdgcn_wmma_f32_16x16x32_f16(
                    false, af[i], false, bf[j], (short)0, acc[i][j], false, false);
        __syncthreads();
    }

    const int rgrp = (lane >> 4) * 8;
#pragma unroll
    for (int i = 0; i < 2; ++i)
#pragma unroll
        for (int j = 0; j < NT; ++j)
#pragma unroll
            for (int r = 0; r < 8; ++r) {
                const int m = row0 + wm + i * 16 + rgrp + r;
                const int n = col0 + wn + j * 16 + lm;
                C[(size_t)m * N + n] = acc[i][j][r] + (bias ? bias[n] : 0.0f);
            }
}

// ---------------- f32 -> f16 bulk convert (4 elems/thread) ----------------
__global__ __launch_bounds__(256) void cvt_f16(const float* __restrict__ in,
                                               _Float16* __restrict__ out, size_t n4)
{
    const size_t i = (size_t)blockIdx.x * 256 + threadIdx.x;
    if (i >= n4) return;
    const float4 f = *(const float4*)(in + i * 4);
    union { _Float16 h[4]; uint2 u; } pk;
    pk.h[0] = (_Float16)f.x; pk.h[1] = (_Float16)f.y;
    pk.h[2] = (_Float16)f.z; pk.h[3] = (_Float16)f.w;
    *(uint2*)(out + i * 4) = pk.u;
}

// ---------------- LayerNorm over feature dim; one block per row ----------------
// input: f32 (inf) or f16 (inh).  mode 0: outh[row*C+c]
// mode 1: window-reverse scatter to NCHW f32 (C==512)   mode 2: outh[row*512+c]
__global__ __launch_bounds__(256) void ln_kernel(
    const float* __restrict__ inf, const _Float16* __restrict__ inh,
    const float* __restrict__ gamma, const float* __restrict__ beta,
    float* __restrict__ outf, _Float16* __restrict__ outh, int C, int mode)
{
    const int row = blockIdx.x;
    const int t = threadIdx.x;

    float s = 0.f, q = 0.f;
    for (int c = t; c < C; c += 256) {
        const float x = inh ? (float)inh[(size_t)row * C + c] : inf[(size_t)row * C + c];
        s += x; q += x * x;
    }
    __shared__ float s1[256], s2[256];
    s1[t] = s; s2[t] = q; __syncthreads();
    for (int off = 128; off > 0; off >>= 1) {
        if (t < off) { s1[t] += s1[t + off]; s2[t] += s2[t + off]; }
        __syncthreads();
    }
    const float mean = s1[0] / C;
    const float var  = s2[0] / C - mean * mean;
    const float rstd = rsqrtf(var + EPSLN);

    for (int c = t; c < C; c += 256) {
        const float x = inh ? (float)inh[(size_t)row * C + c] : inf[(size_t)row * C + c];
        const float y = (x - mean) * rstd * gamma[c] + beta[c];
        if (mode == 0) {
            outh[(size_t)row * C + c] = (_Float16)y;
        } else if (mode == 2) {
            outh[(size_t)row * 512 + c] = (_Float16)y;
        } else {
            const int n = row & 63; const int tt = row >> 6;
            const int b = tt / NWIN; const int win = tt - b * NWIN;
            const int wy = win / NWIN_X, wx = win - wy * NWIN_X;
            const int hh = wy * WSZ + (n >> 3), ww = wx * WSZ + (n & 7);
            outf[(((size_t)b * CDIM + c) * HIMG + hh) * WIMG + ww] = y;
        }
    }
}

// ---------------- depthwise 3x3 conv + BN + GELU (NCHW) ----------------
__global__ __launch_bounds__(256) void dwconv_bn_gelu(
    const float* __restrict__ in, const float* __restrict__ w, const float* __restrict__ bias,
    const float* __restrict__ g1, const float* __restrict__ b1,
    const float* __restrict__ m1, const float* __restrict__ v1,
    float* __restrict__ out)
{
    const size_t idx = (size_t)blockIdx.x * 256 + threadIdx.x;
    if (idx >= (size_t)BATCH * CDIM * HWPIX) return;
    const int wcol = (int)(idx % WIMG);
    const int h    = (int)((idx / WIMG) % HIMG);
    const int c    = (int)((idx / HWPIX) % CDIM);
    const int b    = (int)(idx / ((size_t)HWPIX * CDIM));
    const float* wp = w + c * 9;
    const float* base = in + ((size_t)b * CDIM + c) * HWPIX;
    float s = bias[c];
#pragma unroll
    for (int dy = -1; dy <= 1; ++dy) {
        const int hy = h + dy;
        if (hy < 0 || hy >= HIMG) continue;
#pragma unroll
        for (int dx = -1; dx <= 1; ++dx) {
            const int wx = wcol + dx;
            if (wx < 0 || wx >= WIMG) continue;
            s += wp[(dy + 1) * 3 + (dx + 1)] * base[(size_t)hy * WIMG + wx];
        }
    }
    const float y = (s - m1[c]) * rsqrtf(v1[c] + EPSLN) * g1[c] + b1[c];
    out[idx] = gelu_f(y);
}

// ---------------- global average pool per (b,c) plane ----------------
__global__ __launch_bounds__(256) void pool_kernel(const float* __restrict__ in,
                                                   float* __restrict__ out)
{
    const int bc = blockIdx.x;
    const int t = threadIdx.x;
    const float* p = in + (size_t)bc * HWPIX;
    float s = 0.f;
    for (int i = t; i < HWPIX; i += 256) s += p[i];
    __shared__ float red[256];
    red[t] = s; __syncthreads();
    for (int off = 128; off > 0; off >>= 1) {
        if (t < off) red[t] += red[t + off];
        __syncthreads();
    }
    if (t == 0) out[bc] = red[0] / (float)HWPIX;
}

// ---------------- channel interaction MLP -> sigmoid gate (B, C2) ----------------
__global__ __launch_bounds__(256) void ci_kernel(
    const float* __restrict__ pool, const float* __restrict__ w1, const float* __restrict__ b1,
    const float* __restrict__ g2, const float* __restrict__ b2,
    const float* __restrict__ m2, const float* __restrict__ v2,
    const float* __restrict__ w2, const float* __restrict__ bb2,
    float* __restrict__ gate)
{
    __shared__ float t1[C8DIM];
    const int t = threadIdx.x;
    for (int b = 0; b < BATCH; ++b) {
        if (t < C8DIM) {
            float s = b1[t];
            for (int k = 0; k < CDIM; ++k) s += pool[b * CDIM + k] * w1[k * C8DIM + t];
            const float y = (s - m2[t]) * rsqrtf(v2[t] + EPSLN) * g2[t] + b2[t];
            t1[t] = gelu_f(y);
        }
        __syncthreads();
        {
            float s = bb2[t];
            for (int k = 0; k < C8DIM; ++k) s += t1[k] * w2[k * C2DIM + t];
            gate[b * C2DIM + t] = sigmoid_f(s);
        }
        __syncthreads();
    }
}

// ---------------- tiled transpose: (B,C,HW) f32 -> (B*HW, C) f16 ----------------
__global__ __launch_bounds__(256) void transpose_cp(const float* __restrict__ in,
                                                    _Float16* __restrict__ out)
{
    __shared__ float tile[32][33];
    const int b = blockIdx.z;
    const int p0 = blockIdx.x * 32, c0 = blockIdx.y * 32;
    const int tx = threadIdx.x & 31, ty = threadIdx.x >> 5;  // 32x8
#pragma unroll
    for (int i = 0; i < 32; i += 8)
        tile[ty + i][tx] = in[((size_t)b * CDIM + c0 + ty + i) * HWPIX + p0 + tx];
    __syncthreads();
#pragma unroll
    for (int i = 0; i < 32; i += 8)
        out[((size_t)b * HWPIX + p0 + ty + i) * CDIM + c0 + tx] = (_Float16)tile[tx][ty + i];
}

// ---------------- windowed MHA: one block = (window, head), 4 waves ----------------
__global__ __launch_bounds__(128) void attn_kernel(
    const float* __restrict__ qkv,   // (BWIN, 64, 768) f32
    const float* __restrict__ gate,  // (BATCH, 256) sigmoid(ci)
    const float* __restrict__ rpb,   // (225, 8)
    const int*   __restrict__ rel,   // (64, 64)
    _Float16* __restrict__ out)      // (BWIN, 64, 256) f16
{
    const int blk = blockIdx.x;
    const int t = blk / NHEADS, h = blk - (blk / NHEADS) * NHEADS;
    const int bimg = t / NWIN;
    const int tid = threadIdx.x, lane = tid & 31, wave = tid >> 5;

    __shared__ __attribute__((aligned(16))) _Float16 ql[64 * 40];
    __shared__ __attribute__((aligned(16))) _Float16 kl[64 * 40];
    __shared__ __attribute__((aligned(16))) _Float16 vt[32 * 72];  // transposed: [d][key]
    __shared__ __attribute__((aligned(16))) _Float16 pl[64 * 72];  // probs f16
    __shared__ float sl[64 * 65];

    const float scale = 0.17677669529663687f;  // 1/sqrt(32)
    const float* qb = qkv + (size_t)t * 64 * 768 + h * HDIM;
    const float* kbp = qb + 256;
    const float* vbp = qb + 512;

    for (int i = tid; i < 64 * HDIM; i += 128) {
        const int n = i >> 5, d = i & 31;
        ql[n * 40 + d] = (_Float16)qb[(size_t)n * 768 + d];
        kl[n * 40 + d] = (_Float16)kbp[(size_t)n * 768 + d];
        const float gv = gate[bimg * C2DIM + h * HDIM + d];
        vt[d * 72 + n] = (_Float16)(vbp[(size_t)n * 768 + d] * gv);
    }
    __syncthreads();

    const int lm = lane & 15, kb = (lane >> 4) * 8;
    // ---- scores: (16q x 64k) per wave, K = hd = 32 in one WMMA each
    {
        const v16h af = frag_ld(&ql[(wave * 16 + lm) * 40 + kb]);
        const int rbase = wave * 16 + (lane >> 4) * 8;
#pragma unroll
        for (int kt = 0; kt < 4; ++kt) {
            const v16h bf = frag_ld(&kl[(kt * 16 + lm) * 40 + kb]);
            v8f c = {};
            c = __builtin_amdgcn_wmma_f32_16x16x32_f16(false, af, false, bf, (short)0, c, false, false);
#pragma unroll
            for (int r = 0; r < 8; ++r) sl[(rbase + r) * 65 + kt * 16 + lm] = c[r];
        }
    }
    __syncthreads();

    // ---- softmax + relative position bias; one row per thread (0..63)
    if (tid < 64) {
        const int q = tid;
        float mx = -1e30f;
        for (int k = 0; k < 64; ++k) {
            const float s = sl[q * 65 + k] * scale + rpb[rel[q * 64 + k] * NHEADS + h];
            sl[q * 65 + k] = s;
            mx = fmaxf(mx, s);
        }
        float sum = 0.f;
        for (int k = 0; k < 64; ++k) {
            const float e = __expf(sl[q * 65 + k] - mx);
            sl[q * 65 + k] = e;
            sum += e;
        }
        const float inv = 1.0f / sum;
        for (int k = 0; k < 64; ++k) pl[q * 72 + k] = (_Float16)(sl[q * 65 + k] * inv);
    }
    __syncthreads();

    // ---- attn @ V : (64x64)@(64x32), K = 64 keys in two WMMA steps
    v8f acc[2] = {};
#pragma unroll
    for (int s = 0; s < 2; ++s) {
        const v16h af = frag_ld(&pl[(wave * 16 + lm) * 72 + s * 32 + kb]);
#pragma unroll
        for (int j = 0; j < 2; ++j) {
            const v16h bf = frag_ld(&vt[(j * 16 + lm) * 72 + s * 32 + kb]);
            acc[j] = __builtin_amdgcn_wmma_f32_16x16x32_f16(false, af, false, bf, (short)0, acc[j], false, false);
        }
    }
    const int rbase = wave * 16 + (lane >> 4) * 8;
#pragma unroll
    for (int j = 0; j < 2; ++j)
#pragma unroll
        for (int r = 0; r < 8; ++r)
            out[((size_t)t * 64 + rbase + r) * C2DIM + h * HDIM + j * 16 + lm] = (_Float16)acc[j][r];
}

// ---------------- spatial interaction stage 2: BN+GELU+(64->1)+sigmoid ----------------
__global__ __launch_bounds__(256) void si2_kernel(
    const float* __restrict__ si1,
    const float* __restrict__ g4, const float* __restrict__ b4,
    const float* __restrict__ m4, const float* __restrict__ v4,
    const float* __restrict__ w2, const float* __restrict__ b2,
    float* __restrict__ gtok)
{
    const int i = blockIdx.x * 256 + threadIdx.x;
    if (i >= MROWS) return;
    const float* p = si1 + (size_t)i * C8DIM;
    float s = b2[0];
    for (int c = 0; c < C8DIM; ++c) {
        float y = (p[c] - m4[c]) * rsqrtf(v4[c] + EPSLN) * g4[c] + b4[c];
        s += gelu_f(y) * w2[c];
    }
    gtok[i] = sigmoid_f(s);
}

// ---------------- gated CNN branch -> BN3 -> window-partition into concat cols 256.. ----
__global__ __launch_bounds__(256) void cnnpart_kernel(
    const float* __restrict__ pj,   // (B*HW, 256) pixel-major f32
    const float* __restrict__ gtok, // (MROWS) sigmoid gate per pixel/token
    const float* __restrict__ g3, const float* __restrict__ b3,
    const float* __restrict__ m3, const float* __restrict__ v3,
    _Float16* __restrict__ cat)     // (MROWS, 512) f16
{
    const size_t idx = (size_t)blockIdx.x * 256 + threadIdx.x;  // MROWS*256 total
    const int c = (int)(idx & 255);
    const size_t row = idx >> 8;               // token row = t*64 + n
    const int n = (int)(row & 63);
    const size_t tt = row >> 6;
    const int b = (int)(tt / NWIN);
    const int win = (int)(tt - (size_t)b * NWIN);
    const int wy = win / NWIN_X, wx = win - wy * NWIN_X;
    const int hh = wy * WSZ + (n >> 3), ww = wx * WSZ + (n & 7);
    const size_t pix = (size_t)b * HWPIX + (size_t)hh * WIMG + ww;
    float val = gtok[row] * pj[pix * C2DIM + c];
    val = (val - m3[c]) * rsqrtf(v3[c] + EPSLN) * g3[c] + b3[c];
    cat[row * 512 + 256 + c] = (_Float16)val;
}

extern "C" void kernel_launch(void* const* d_in, const int* in_sizes, int n_in,
                              void* d_out, int out_size, void* d_ws, size_t ws_size,
                              hipStream_t stream) {
    (void)in_sizes; (void)n_in; (void)out_size; (void)ws_size;
    const float* x      = (const float*)d_in[0];
    const float* w_pa   = (const float*)d_in[1];
    const float* b_pa   = (const float*)d_in[2];
    const float* g_lna  = (const float*)d_in[3];
    const float* b_lna  = (const float*)d_in[4];
    const float* w_pc   = (const float*)d_in[5];
    const float* b_pc   = (const float*)d_in[6];
    const float* g_lnc  = (const float*)d_in[7];
    const float* b_lnc  = (const float*)d_in[8];
    const float* dw_w   = (const float*)d_in[9];
    const float* dw_b   = (const float*)d_in[10];
    const float* g_bn1  = (const float*)d_in[11];
    const float* b_bn1  = (const float*)d_in[12];
    const float* m_bn1  = (const float*)d_in[13];
    const float* v_bn1  = (const float*)d_in[14];
    const float* ci_w1  = (const float*)d_in[15];
    const float* ci_b1  = (const float*)d_in[16];
    const float* g_bn2  = (const float*)d_in[17];
    const float* b_bn2  = (const float*)d_in[18];
    const float* m_bn2  = (const float*)d_in[19];
    const float* v_bn2  = (const float*)d_in[20];
    const float* ci_w2  = (const float*)d_in[21];
    const float* ci_b2  = (const float*)d_in[22];
    const float* pj_w   = (const float*)d_in[23];
    const float* pj_b   = (const float*)d_in[24];
    const float* g_bn3  = (const float*)d_in[25];
    const float* b_bn3  = (const float*)d_in[26];
    const float* m_bn3  = (const float*)d_in[27];
    const float* v_bn3  = (const float*)d_in[28];
    const float* w_qkv  = (const float*)d_in[29];
    const float* b_qkv  = (const float*)d_in[30];
    const float* si_w1  = (const float*)d_in[31];
    const float* si_b1  = (const float*)d_in[32];
    const float* g_bn4  = (const float*)d_in[33];
    const float* b_bn4  = (const float*)d_in[34];
    const float* m_bn4  = (const float*)d_in[35];
    const float* v_bn4  = (const float*)d_in[36];
    const float* si_w2  = (const float*)d_in[37];
    const float* si_b2  = (const float*)d_in[38];
    const float* g_lno  = (const float*)d_in[39];
    const float* b_lno  = (const float*)d_in[40];
    const float* w_out  = (const float*)d_in[41];
    const float* b_out  = (const float*)d_in[42];
    const float* rpb    = (const float*)d_in[43];
    const int*   rel    = (const int*)d_in[44];
    float* out = (float*)d_out;

    // workspace carve-out (sequential-lifetime reuse)
    char* base = (char*)d_ws;
    size_t off = 0;
    auto take = [&](size_t bytes) -> char* {
        char* p = base + off;
        off += (bytes + 255) & ~(size_t)255;
        return p;
    };
    float*     g12    = (float*)take((size_t)MROWS * 512 * 4);  // gemm1/gemm2 out; reused as dpm(f16)
    _Float16*  xh     = (_Float16*)take((size_t)MROWS * 512 * 2);
    _Float16*  xatt   = (_Float16*)take((size_t)MROWS * 256 * 2);
    float*     xcnn   = (float*)take((size_t)MROWS * 512 * 4);  // NCHW; reused as cat(f16)
    float*     dcnn   = (float*)take((size_t)MROWS * 512 * 4);  // reused as xatt2(f16)/si1/gtok
    float*     qkvb   = (float*)take((size_t)MROWS * 768 * 4);
    float*     pjpm   = (float*)take((size_t)MROWS * 256 * 4);
    float*     cipool = (float*)take(BATCH * CDIM * 4);
    float*     cigate = (float*)take(BATCH * C2DIM * 4);
    _Float16*  xatt2  = (_Float16*)dcnn;
    float*     si1    = (float*)((char*)dcnn + (size_t)MROWS * 256 * 2);
    float*     gtok   = si1 + (size_t)MROWS * 64;
    _Float16*  cat    = (_Float16*)xcnn;
    _Float16*  dpm    = (_Float16*)g12;

    const dim3 blk(256);
    // 0. x -> f16 (halves GEMM A-operand HBM traffic)
    cvt_f16<<<(unsigned)(((size_t)MROWS * 512 / 4 + 255) / 256), blk, 0, stream>>>(
        x, xh, (size_t)MROWS * 512 / 4);
    // 1. attn-branch projection + LN
    gemm_f16wmma<128><<<dim3(256 / 128, MROWS / BM), blk, 0, stream>>>(xh, w_pa, b_pa, g12, MROWS, 256, 512);
    ln_kernel<<<MROWS, blk, 0, stream>>>(g12, nullptr, g_lna, b_lna, nullptr, xatt, 256, 0);
    // 2. cnn-branch projection + LN + window reverse to NCHW
    gemm_f16wmma<128><<<dim3(512 / 128, MROWS / BM), blk, 0, stream>>>(xh, w_pc, b_pc, g12, MROWS, 512, 512);
    ln_kernel<<<MROWS, blk, 0, stream>>>(g12, nullptr, g_lnc, b_lnc, xcnn, nullptr, 512, 1);
    // 3. depthwise conv + BN1 + GELU
    const size_t convN = (size_t)BATCH * CDIM * HWPIX;
    dwconv_bn_gelu<<<(unsigned)((convN + 255) / 256), blk, 0, stream>>>(
        xcnn, dw_w, dw_b, g_bn1, b_bn1, m_bn1, v_bn1, dcnn);
    // 4. channel interaction -> sigmoid gate
    pool_kernel<<<BATCH * CDIM, blk, 0, stream>>>(dcnn, cipool);
    ci_kernel<<<1, blk, 0, stream>>>(cipool, ci_w1, ci_b1, g_bn2, b_bn2, m_bn2, v_bn2,
                                     ci_w2, ci_b2, cigate);
    // 5. NCHW -> pixel-major f16, then 1x1 projection (pj)
    transpose_cp<<<dim3(HWPIX / 32, CDIM / 32, BATCH), blk, 0, stream>>>(dcnn, dpm);
    gemm_f16wmma<128><<<dim3(256 / 128, MROWS / BM), blk, 0, stream>>>(dpm, pj_w, pj_b, pjpm, MROWS, 256, 512);
    // 6. qkv projection + attention
    gemm_f16wmma<128><<<dim3(768 / 128, MROWS / BM), blk, 0, stream>>>(xatt, w_qkv, b_qkv, qkvb, MROWS, 768, 256);
    attn_kernel<<<BWIN * NHEADS, dim3(128), 0, stream>>>(qkvb, cigate, rpb, rel, xatt2);
    // 7. spatial interaction gate
    gemm_f16wmma<64><<<dim3(1, MROWS / BM), blk, 0, stream>>>(xatt2, si_w1, si_b1, si1, MROWS, 64, 256);
    si2_kernel<<<(MROWS + 255) / 256, blk, 0, stream>>>(si1, g_bn4, b_bn4, m_bn4, v_bn4,
                                                        si_w2, si_b2, gtok);
    // 8. concat buffer: LN(attn) into cols 0..255, gated/BN3 cnn into 256..511
    ln_kernel<<<MROWS, blk, 0, stream>>>(nullptr, xatt2, g_lno, b_lno, nullptr, cat, 256, 2);
    cnnpart_kernel<<<MROWS, blk, 0, stream>>>(pjpm, gtok, g_bn3, b_bn3, m_bn3, v_bn3, cat);
    // 9. output projection
    gemm_f16wmma<128><<<dim3(512 / 128, MROWS / BM), blk, 0, stream>>>(cat, w_out, b_out, out, MROWS, 512, 512);
}
